// AdaptGCN_48601849922155
// MI455X (gfx1250) — matrix operand order repaired and checked by reading, not compiled, not run on hardware
//
#include <hip/hip_runtime.h>
#include <hip/hip_bf16.h>

// ---------------------------------------------------------------------------
// AdaptGCN on MI455X (gfx1250), wave32, WMMA bf16 path.
//   N = 4096, C_IN = 4096, HID = 64, C_OUT = 64
// ada = x @ W_lin1^T only feeds a nonzero mask -> bf16 WMMA w/ f32 accum is
// safe. x is bf16-converted once; W_lin1/W1/W2 are bf16-converted AND
// pre-transposed to k-major once, so every WMMA kernel's K-loop staging is
// pure b128 global->LDS copies (no converts, no scalar DS transpose stores).
// ---------------------------------------------------------------------------

typedef __attribute__((ext_vector_type(16))) __bf16          v16bf;
typedef __attribute__((ext_vector_type(8)))  float           v8f;
typedef __attribute__((ext_vector_type(8)))  unsigned short  ush8;

union BfFrag { v16bf v; ush8 h[2]; };

#define NN    4096
#define HIDN  64
#define TK    32
#define APAD  8          // row strides stay multiples of 8 ushorts (16B align)
#define BPAD  8

__device__ __forceinline__ unsigned short f32_to_bf16_rne(float f) {
    unsigned int u = __float_as_uint(f);
    unsigned int r = u + 0x7FFFu + ((u >> 16) & 1u);
    return (unsigned short)(r >> 16);
}

// ---------------------------------------------------------------------------
// Kernel 0a: streaming f32 -> bf16 conversion (8 elements / thread)
// ---------------------------------------------------------------------------
__global__ __launch_bounds__(256) void to_bf16_kernel(
    const float* __restrict__ src, unsigned short* __restrict__ dst)
{
    const int i = blockIdx.x * 256 + threadIdx.x;
    const float4* s = (const float4*)src + (size_t)i * 2;
    const float4 u = s[0], v = s[1];
    ush8 o;
    o[0] = f32_to_bf16_rne(u.x); o[1] = f32_to_bf16_rne(u.y);
    o[2] = f32_to_bf16_rne(u.z); o[3] = f32_to_bf16_rne(u.w);
    o[4] = f32_to_bf16_rne(v.x); o[5] = f32_to_bf16_rne(v.y);
    o[6] = f32_to_bf16_rne(v.z); o[7] = f32_to_bf16_rne(v.w);
    ((ush8*)dst)[i] = o;
}

// ---------------------------------------------------------------------------
// Kernel 0b: dst[c][r] = bf16(src[r][c]).  R, C multiples of 64.
// 64x64 tile via padded LDS; both global sides coalesced (one-time pass).
// ---------------------------------------------------------------------------
__global__ __launch_bounds__(256) void transpose_bf16_kernel(
    const float* __restrict__ src, unsigned short* __restrict__ dst,
    int R, int C)
{
    __shared__ unsigned short T[64][64 + BPAD];   // 144B rows (16B multiple)
    const int tid = threadIdx.x;
    const int c0  = blockIdx.x * 64;
    const int r0  = blockIdx.y * 64;

    const int lr = tid >> 2, lc = (tid & 3) * 16;       // 16 f32 per thread
    const float4* s = (const float4*)(src + (size_t)(r0 + lr) * C + c0 + lc);
    #pragma unroll
    for (int q = 0; q < 4; ++q) {
        const float4 u = s[q];
        T[lc + q * 4 + 0][lr] = f32_to_bf16_rne(u.x);
        T[lc + q * 4 + 1][lr] = f32_to_bf16_rne(u.y);
        T[lc + q * 4 + 2][lr] = f32_to_bf16_rne(u.z);
        T[lc + q * 4 + 3][lr] = f32_to_bf16_rne(u.w);
    }
    __syncthreads();

    const int oc = tid >> 2, orr = (tid & 3) * 16;      // 16 bf16 per thread
    ush8*       d = (ush8*)(dst + (size_t)(c0 + oc) * R + r0 + orr);
    const ush8* t = (const ush8*)&T[oc][orr];
    d[0] = t[0];
    d[1] = t[1];
}

// ---------------------------------------------------------------------------
// Kernel 1: ada = x @ W_lin1^T + b_lin1 ; maskT[dst][src] = (ada[src][dst]!=0)
// 256 threads = 8 waves (4M x 2N); block tile 128(M=src) x 64(N=dst);
// each wave computes 32x32 with 4 WMMAs per K-step (2 A-frags x 2 B-frags).
// B operand (wT) is already k-major bf16 -> staging is pure b128 copies.
// ---------------------------------------------------------------------------
__global__ __launch_bounds__(256) void ada_mask_kernel(
    const unsigned short* __restrict__ xb,     // [4096,4096] bf16 (row-major)
    const unsigned short* __restrict__ wT,     // [4096 k][4096 dst] bf16
    const float* __restrict__ bl1,             // [4096]
    unsigned short* __restrict__ maskT)        // [dst][src] bf16 {0,1}
{
    __shared__ unsigned short As[128][TK + APAD];   // 80B rows
    __shared__ unsigned short Bt[TK][64 + BPAD];    // 144B rows
    __shared__ unsigned short Ct[64][128];          // transposed mask tile

    const int tid   = threadIdx.x;
    const int M0    = blockIdx.x * 128;    // src
    const int N0    = blockIdx.y * 64;     // dst
    const int lane  = tid & 31;
    const int wave  = tid >> 5;
    const int wm    = (wave & 3) * 32;
    const int wn    = (wave >> 2) * 32;
    const int mrow  = lane & 15;
    const int khalf = lane >> 4;

    const int a_r = tid >> 1, a_c = (tid & 1) * 16;  // A stage: 2x ush8
    const int y_k = tid >> 3, y_n = (tid & 7) * 8;   // B stage: 1x ush8

    v8f acc00 = {}, acc01 = {}, acc10 = {}, acc11 = {};

    const unsigned short* srcA = xb + (size_t)(M0 + a_r) * NN + a_c;
    const unsigned short* srcB = wT + (size_t)y_k * NN + N0 + y_n;

    for (int k0 = 0; k0 < NN; k0 += TK) {
        __syncthreads();
        *(ush8*)&As[a_r][a_c]     = *(const ush8*)(srcA + k0);
        *(ush8*)&As[a_r][a_c + 8] = *(const ush8*)(srcA + k0 + 8);
        *(ush8*)&Bt[y_k][y_n]     = *(const ush8*)(srcB + (size_t)k0 * NN);
        if (k0 + TK < NN) {   // prefetch next K tile (global_prefetch_b8)
            __builtin_prefetch((const void*)(srcA + k0 + TK), 0, 1);
            __builtin_prefetch((const void*)(srcB + (size_t)(k0 + TK) * NN), 0, 1);
        }
        __syncthreads();

        BfFrag a0, a1, b0, b1;
        const ush8* ap0 = (const ush8*)&As[wm + mrow][0];
        a0.h[0] = ap0[khalf];      a0.h[1] = ap0[2 + khalf];
        const ush8* ap1 = (const ush8*)&As[wm + 16 + mrow][0];
        a1.h[0] = ap1[khalf];      a1.h[1] = ap1[2 + khalf];
        #pragma unroll
        for (int j = 0; j < 8; ++j) {
            b0.h[0][j] = Bt[8 * khalf + j][wn + mrow];      b0.h[1][j] = Bt[16 + 8 * khalf + j][wn + mrow];
            b1.h[0][j] = Bt[8 * khalf + j][wn + 16 + mrow]; b1.h[1][j] = Bt[16 + 8 * khalf + j][wn + 16 + mrow];
        }

        acc00 = __builtin_amdgcn_wmma_f32_16x16x32_bf16(false, a0.v, false, b0.v,
                                                        (short)0, acc00, false, false);
        acc01 = __builtin_amdgcn_wmma_f32_16x16x32_bf16(false, a0.v, false, b1.v,
                                                        (short)0, acc01, false, false);
        acc10 = __builtin_amdgcn_wmma_f32_16x16x32_bf16(false, a1.v, false, b0.v,
                                                        (short)0, acc10, false, false);
        acc11 = __builtin_amdgcn_wmma_f32_16x16x32_bf16(false, a1.v, false, b1.v,
                                                        (short)0, acc11, false, false);
    }

    __syncthreads();
    {   // epilogue: bias over dst, nonzero test, transposed tile in LDS
        const float bn0 = bl1[N0 + wn + mrow];
        const float bn1 = bl1[N0 + wn + mrow + 16];
        #pragma unroll
        for (int r = 0; r < 8; ++r) {
            const int mB = wm + r + 8 * khalf;
            const int n0 = wn + mrow, n1 = wn + 16 + mrow;
            Ct[n0][mB]      = ((acc00[r] + bn0) != 0.0f) ? (unsigned short)0x3F80u : 0u;
            Ct[n1][mB]      = ((acc01[r] + bn1) != 0.0f) ? (unsigned short)0x3F80u : 0u;
            Ct[n0][mB + 16] = ((acc10[r] + bn0) != 0.0f) ? (unsigned short)0x3F80u : 0u;
            Ct[n1][mB + 16] = ((acc11[r] + bn1) != 0.0f) ? (unsigned short)0x3F80u : 0u;
        }
    }
    __syncthreads();
    {   // coalesced store: maskT[(N0+n)][M0 + m], 32 elems / thread
        const int n  = tid >> 2;
        const int m0 = (tid & 3) * 32;
        ush8*       dst  = (ush8*)(maskT + (size_t)(N0 + n) * NN + M0 + m0);
        const ush8* srcp = (const ush8*)&Ct[n][m0];
        dst[0] = srcp[0];
        dst[1] = srcp[1];
        dst[2] = srcp[2];
        dst[3] = srcp[3];
    }
}

// ---------------------------------------------------------------------------
// Kernel 2: deg[t] = 1 + rowsum(maskT[t]); dis = rsqrt(deg); invdeg = 1/deg
// ---------------------------------------------------------------------------
__global__ __launch_bounds__(256) void degree_kernel(
    const unsigned short* __restrict__ maskT,
    float* __restrict__ dis, float* __restrict__ invdeg)
{
    __shared__ float red[256];
    const int t = blockIdx.x;
    const ush8* row = (const ush8*)(maskT + (size_t)t * NN);   // 512 chunks
    int cnt = 0;
    for (int i = threadIdx.x; i < NN / 8; i += 256) {
        const ush8 m = row[i];
        #pragma unroll
        for (int j = 0; j < 8; ++j) cnt += (m[j] != 0) ? 1 : 0;
    }
    red[threadIdx.x] = (float)cnt;
    __syncthreads();
    for (int s = 128; s > 0; s >>= 1) {
        if (threadIdx.x < s) red[threadIdx.x] += red[threadIdx.x + s];
        __syncthreads();
    }
    if (threadIdx.x == 0) {
        const float d = 1.0f + red[0];
        dis[t]    = rsqrtf(d);
        invdeg[t] = 1.0f / d;
    }
}

// ---------------------------------------------------------------------------
// Kernel 3/6: hp = A @ W^T  (A:[M,K] bf16, WT:[K,64] bf16 k-major, hp f32)
// ---------------------------------------------------------------------------
__global__ __launch_bounds__(256) void gemm_nt64_kernel(
    const unsigned short* __restrict__ A, const unsigned short* __restrict__ WT,
    float* __restrict__ C, int K)
{
    __shared__ unsigned short As[64][TK + APAD];
    __shared__ unsigned short Bt[TK][64 + BPAD];

    const int tid   = threadIdx.x;
    const int M0    = blockIdx.x * 64;
    const int lane  = tid & 31;
    const int wave  = tid >> 5;
    const int wm    = (wave & 3) * 16;
    const int wn    = (wave >> 2) * 32;
    const int mrow  = lane & 15;
    const int khalf = lane >> 4;

    const int a_r = tid >> 2, a_c = (tid & 3) * 8;
    const int y_k = tid >> 3, y_n = (tid & 7) * 8;

    v8f acc0 = {}; v8f acc1 = {};

    for (int k0 = 0; k0 < K; k0 += TK) {
        __syncthreads();
        *(ush8*)&As[a_r][a_c] = *(const ush8*)(A  + (size_t)(M0 + a_r) * K + k0 + a_c);
        *(ush8*)&Bt[y_k][y_n] = *(const ush8*)(WT + (size_t)(k0 + y_k) * HIDN + y_n);
        __syncthreads();

        BfFrag a, b0, b1;
        const ush8* ap  = (const ush8*)&As[wm + mrow][0];
        a.h[0] = ap[khalf];
        a.h[1] = ap[2 + khalf];
        #pragma unroll
        for (int j = 0; j < 8; ++j) {
            b0.h[0][j] = Bt[8 * khalf + j][wn + mrow];      b0.h[1][j] = Bt[16 + 8 * khalf + j][wn + mrow];
            b1.h[0][j] = Bt[8 * khalf + j][wn + 16 + mrow]; b1.h[1][j] = Bt[16 + 8 * khalf + j][wn + 16 + mrow];
        }

        acc0 = __builtin_amdgcn_wmma_f32_16x16x32_bf16(false, a.v, false, b0.v,
                                                       (short)0, acc0, false, false);
        acc1 = __builtin_amdgcn_wmma_f32_16x16x32_bf16(false, a.v, false, b1.v,
                                                       (short)0, acc1, false, false);
    }

    #pragma unroll
    for (int r = 0; r < 8; ++r) {
        const int m = M0 + wm + r + 8 * khalf;
        const int f = wn + mrow;
        C[(size_t)m * HIDN + f]      = acc0[r];
        C[(size_t)m * HIDN + f + 16] = acc1[r];
    }
}

// ---------------------------------------------------------------------------
// Kernel 4/7: Y[n][f] = bf16(dis[n] * hp[n][f])
// ---------------------------------------------------------------------------
__global__ __launch_bounds__(256) void scale_kernel(
    const float* __restrict__ hp, const float* __restrict__ dis,
    unsigned short* __restrict__ Y)
{
    const int i = blockIdx.x * 256 + threadIdx.x;   // 4096*64 total
    Y[i] = f32_to_bf16_rne(dis[i >> 6] * hp[i]);
}

// ---------------------------------------------------------------------------
// Kernel 5/8: H[t][f] = dis[t]*(maskT[t,:] @ Y[:,f]) + hp[t][f]*invdeg[t] + b[f]
// optionally ReLU; also emits a fused bf16 copy (feeds the next K=64 GEMM).
// ---------------------------------------------------------------------------
__global__ __launch_bounds__(256) void propagate_kernel(
    const unsigned short* __restrict__ maskT,   // [4096,4096] bf16
    const unsigned short* __restrict__ Y,       // [4096,64] bf16 (k-major)
    const float* __restrict__ dis, const float* __restrict__ invdeg,
    const float* __restrict__ hp,               // [4096,64] f32
    const float* __restrict__ bias,             // [64]
    float* __restrict__ Hout,                   // [4096,64] f32
    unsigned short* __restrict__ HbfOut,        // [4096,64] bf16
    int relu)
{
    __shared__ unsigned short As[64][TK + APAD];
    __shared__ unsigned short Bt[TK][64 + BPAD];

    const int tid   = threadIdx.x;
    const int M0    = blockIdx.x * 64;
    const int lane  = tid & 31;
    const int wave  = tid >> 5;
    const int wm    = (wave & 3) * 16;
    const int wn    = (wave >> 2) * 32;
    const int mrow  = lane & 15;
    const int khalf = lane >> 4;

    const int a_r = tid >> 2, a_c = (tid & 3) * 8;   // 16B vector copies
    const int y_k = tid >> 3, y_n = (tid & 7) * 8;

    v8f acc0 = {}; v8f acc1 = {};

    const unsigned short* srcA = maskT + (size_t)(M0 + a_r) * NN + a_c;

    for (int k0 = 0; k0 < NN; k0 += TK) {
        __syncthreads();
        *(ush8*)&As[a_r][a_c] = *(const ush8*)(srcA + k0);
        *(ush8*)&Bt[y_k][y_n] = *(const ush8*)(Y + (size_t)(k0 + y_k) * HIDN + y_n);
        if (k0 + TK < NN)
            __builtin_prefetch((const void*)(srcA + k0 + TK), 0, 1);
        __syncthreads();

        BfFrag a, b0, b1;
        const ush8* ap  = (const ush8*)&As[wm + mrow][0];
        a.h[0] = ap[khalf];
        a.h[1] = ap[2 + khalf];
        #pragma unroll
        for (int j = 0; j < 8; ++j) {
            b0.h[0][j] = Bt[8 * khalf + j][wn + mrow];      b0.h[1][j] = Bt[16 + 8 * khalf + j][wn + mrow];
            b1.h[0][j] = Bt[8 * khalf + j][wn + 16 + mrow]; b1.h[1][j] = Bt[16 + 8 * khalf + j][wn + 16 + mrow];
        }

        acc0 = __builtin_amdgcn_wmma_f32_16x16x32_bf16(false, a.v, false, b0.v,
                                                       (short)0, acc0, false, false);
        acc1 = __builtin_amdgcn_wmma_f32_16x16x32_bf16(false, a.v, false, b1.v,
                                                       (short)0, acc1, false, false);
    }

    const float bf0 = bias[wn + mrow];
    const float bf1 = bias[wn + mrow + 16];
    #pragma unroll
    for (int r = 0; r < 8; ++r) {
        const int t = M0 + wm + r + 8 * khalf;
        const int f = wn + mrow;
        const float ds = dis[t], iv = invdeg[t];
        float v0 = ds * acc0[r] + hp[(size_t)t * HIDN + f]      * iv + bf0;
        float v1 = ds * acc1[r] + hp[(size_t)t * HIDN + f + 16] * iv + bf1;
        if (relu) { v0 = fmaxf(v0, 0.0f); v1 = fmaxf(v1, 0.0f); }
        Hout[(size_t)t * HIDN + f]        = v0;
        Hout[(size_t)t * HIDN + f + 16]   = v1;
        HbfOut[(size_t)t * HIDN + f]      = f32_to_bf16_rne(v0);
        HbfOut[(size_t)t * HIDN + f + 16] = f32_to_bf16_rne(v1);
    }
}

// ---------------------------------------------------------------------------
// Kernel 9: out[c] = W_lin2[c,:] . h2.flatten() + b_lin2[c]   (64 rows)
// ---------------------------------------------------------------------------
__global__ __launch_bounds__(256) void final_matvec_kernel(
    const float* __restrict__ Wl2, const float* __restrict__ h2,
    const float* __restrict__ bl2, float* __restrict__ out)
{
    __shared__ float red[256];
    const int c = blockIdx.x;
    const float* wrow = Wl2 + (size_t)c * (NN * HIDN);
    float s = 0.0f;
    for (int i = threadIdx.x; i < NN * HIDN; i += 256) s += wrow[i] * h2[i];
    red[threadIdx.x] = s;
    __syncthreads();
    for (int st = 128; st > 0; st >>= 1) {
        if (threadIdx.x < st) red[threadIdx.x] += red[threadIdx.x + st];
        __syncthreads();
    }
    if (threadIdx.x == 0) out[c] = red[0] + bl2[c];
}

// ---------------------------------------------------------------------------
extern "C" void kernel_launch(void* const* d_in, const int* in_sizes, int n_in,
                              void* d_out, int out_size, void* d_ws, size_t ws_size,
                              hipStream_t stream) {
    const float* x      = (const float*)d_in[0];
    const float* W_lin1 = (const float*)d_in[1];
    const float* b_lin1 = (const float*)d_in[2];
    const float* W1     = (const float*)d_in[3];
    const float* b1     = (const float*)d_in[4];
    const float* W2     = (const float*)d_in[5];
    const float* b2     = (const float*)d_in[6];
    const float* W_lin2 = (const float*)d_in[7];
    const float* b_lin2 = (const float*)d_in[8];
    float* out = (float*)d_out;

    // workspace layout (~101 MB)
    char* ws = (char*)d_ws;
    unsigned short* maskT = (unsigned short*)(ws);                        // 32 MB
    unsigned short* x_bf  = (unsigned short*)(ws + (size_t)33554432);     // 32 MB
    unsigned short* wT_bf = (unsigned short*)(ws + (size_t)67108864);     // 32 MB (k-major)
    unsigned short* Y     = (unsigned short*)(ws + (size_t)100663296);    // 512 KB
    float* hp             = (float*)(ws + (size_t)101187584);             // 1 MB
    float* h1             = (float*)(ws + (size_t)102236160);             // 1 MB
    float* h2             = (float*)(ws + (size_t)103284736);             // 1 MB
    unsigned short* h1_bf = (unsigned short*)(ws + (size_t)104333312);    // 512 KB
    unsigned short* W1T   = (unsigned short*)(ws + (size_t)104857600);    // 512 KB [4096,64]
    unsigned short* W2T   = (unsigned short*)(ws + (size_t)105381888);    // 8 KB   [64,64]
    float* dis            = (float*)(ws + (size_t)105398272);             // 16 KB
    float* invdeg         = (float*)(ws + (size_t)105414656);             // 16 KB

    const int cvt_blocks = (NN * NN) / (256 * 8);     // 8192

    // 0) one-time precision/layout transforms
    to_bf16_kernel<<<cvt_blocks, 256, 0, stream>>>(x, x_bf);
    transpose_bf16_kernel<<<dim3(NN / 64, NN / 64), 256, 0, stream>>>(W_lin1, wT_bf, NN, NN);
    transpose_bf16_kernel<<<dim3(NN / 64, 1), 256, 0, stream>>>(W1, W1T, HIDN, NN);
    transpose_bf16_kernel<<<dim3(1, 1), 256, 0, stream>>>(W2, W2T, HIDN, HIDN);
    // 1) ada GEMM -> nonzero mask (transposed, bf16 {0,1})
    ada_mask_kernel<<<dim3(NN / 128, NN / 64), 256, 0, stream>>>(x_bf, wT_bf, b_lin1, maskT);
    // 2) degrees
    degree_kernel<<<NN, 256, 0, stream>>>(maskT, dis, invdeg);
    // 3) hp1 = x @ W1^T
    gemm_nt64_kernel<<<NN / 64, 256, 0, stream>>>(x_bf, W1T, hp, NN);
    // 4) Y = bf16(dis * hp1)
    scale_kernel<<<(NN * HIDN) / 256, 256, 0, stream>>>(hp, dis, Y);
    // 5) h1 = relu(dis*(maskT@Y) + hp1*invdeg + b1)   (+ bf16 copy)
    propagate_kernel<<<NN / 64, 256, 0, stream>>>(maskT, Y, dis, invdeg, hp, b1, h1, h1_bf, 1);
    // 6) hp2 = h1 @ W2^T
    gemm_nt64_kernel<<<NN / 64, 256, 0, stream>>>(h1_bf, W2T, hp, HIDN);
    // 7) Y = bf16(dis * hp2)
    scale_kernel<<<(NN * HIDN) / 256, 256, 0, stream>>>(hp, dis, Y);
    // 8) h2 = dis*(maskT@Y) + hp2*invdeg + b2   (bf16 copy unused scratch)
    propagate_kernel<<<NN / 64, 256, 0, stream>>>(maskT, Y, dis, invdeg, hp, b2, h2, h1_bf, 0);
    // 9) out = W_lin2 @ h2.flatten() + b_lin2
    final_matvec_kernel<<<HIDN, 256, 0, stream>>>(W_lin2, h2, b_lin2, out);
}
